// CoeusMemoryLayer_29154238005991
// MI455X (gfx1250) — matrix operand order repaired and checked by reading, not compiled, not run on hardware
//
#include <hip/hip_runtime.h>
#include <math.h>

#define T_   2048
#define C_   1024
#define H_   16
#define HKV_ 4
#define DH_  64
#define WIN_ 128

typedef __attribute__((ext_vector_type(16))) __bf16 v16bf;
typedef __attribute__((ext_vector_type(8)))  float  v8f;
typedef __attribute__((ext_vector_type(8)))  unsigned short v8us;   // 16 bytes
typedef __attribute__((ext_vector_type(4)))  unsigned int   v4ui;
typedef __attribute__((ext_vector_type(8)))  int            v8i_t;
typedef __attribute__((ext_vector_type(4)))  int            v4i_t;

union Frag { v16bf v; v8us h[2]; unsigned short u[16]; };

__device__ __forceinline__ unsigned short f2bf(float f) {
    unsigned int u = __float_as_uint(f);
    unsigned int r = u + 0x7FFFu + ((u >> 16) & 1u);
    return (unsigned short)(r >> 16);
}

__device__ __forceinline__ v8f wmma_bf16(v16bf a, v16bf b, v8f c) {
    return __builtin_amdgcn_wmma_f32_16x16x32_bf16(false, a, false, b, (short)0, c, false, false);
}

// ---- CDNA5 async memory->LDS copy (16B per lane), tracked by ASYNCcnt ----
__device__ __forceinline__ void async_cp16(unsigned short* ldst, const unsigned short* gsrc) {
    unsigned int lds_addr = (unsigned int)(unsigned long long)(void*)ldst;  // LDS aperture: low 32 bits
    asm volatile("global_load_async_to_lds_b128 %0, %1, off"
                 :: "v"(lds_addr), "v"(gsrc)
                 : "memory");
}
template<int N>
__device__ __forceinline__ void wait_async() {
    asm volatile("s_wait_asynccnt %0" :: "n"(N) : "memory");
}
__device__ __forceinline__ void wait_lds() {
    asm volatile("s_wait_dscnt 0x0" ::: "memory");
}

// ---- CDNA5 Tensor Data Mover: one instruction moves a 2D bf16 tile to LDS ----
// D# per ISA 8.3/8.4: group0 = {count|flags, lds_addr, global_addr[56:0], type=2},
// group1 = {data_size=2B, tensor_dim0/1, tile_dim0/1, dim0_stride}. 2D -> groups 2/3 zero.
__device__ __forceinline__ void tdm_load_2d(unsigned short* lds, const unsigned short* gsrc,
                                            unsigned tile_w, unsigned tile_h, unsigned gstride) {
    unsigned long long ga = (unsigned long long)gsrc;
    v4ui g0;
    g0[0] = 1u;                                                   // count=1, user descriptor
    g0[1] = (unsigned int)(unsigned long long)(void*)lds;         // lds byte address
    g0[2] = (unsigned int)ga;                                     // global_addr[31:0]
    g0[3] = (unsigned int)((ga >> 32) & 0x01FFFFFFull) | 0x80000000u;  // addr[56:32] | type=2
    v8i_t g1;
    g1[0] = 0x10000;                      // workgroup_mask=0, data_size=1 (2 bytes)
    g1[1] = (int)(tile_w << 16);          // tensor_dim0[15:0] (bits 63:48)
    g1[2] = (int)(tile_h << 16);          // tensor_dim0 hi=0 | tensor_dim1[15:0]
    g1[3] = (int)(tile_w << 16);          // tensor_dim1 hi=0 | tile_dim0
    g1[4] = (int)tile_h;                  // tile_dim1 | tile_dim2=0
    g1[5] = (int)gstride;                 // tensor_dim0_stride[31:0] (elems)
    g1[6] = 0;                            // stride hi | dim1_stride lo
    g1[7] = 0;
    v4i_t z = {};
#if defined(__clang_major__) && __clang_major__ >= 23
    v8i_t z8 = {};
    __builtin_amdgcn_tensor_load_to_lds(g0, g1, z, z, z8, 0);
#else
    __builtin_amdgcn_tensor_load_to_lds(g0, g1, z, z, 0);
#endif
}
template<int N>
__device__ __forceinline__ void wait_tensor() {
    __builtin_amdgcn_s_wait_tensorcnt(N);
}

// Stage rows x cols(u16) tile (global row stride gld) into contiguous LDS via
// per-lane async DMA: CNT 16B vectors per thread, CV vectors per row.
template<int CNT, int CV>
__device__ __forceinline__ void async_stage_tile(unsigned short* dst, const unsigned short* src,
                                                 int gld, int tid, int nth) {
#pragma unroll
    for (int k = 0; k < CNT; ++k) {
        int i = tid + k * nth;
        int r = i / CV, c8 = (i % CV) * 8;
        async_cp16(dst + i * 8, src + (size_t)r * gld + c8);
    }
}

// A fragment: 16x32 bf16 row-major tile, row stride ld (elems) -> 2x ds_load_b128.
__device__ __forceinline__ v16bf load_a_frag(const unsigned short* s, int ld) {
    int lane = threadIdx.x & 31;
    int m = lane & 15, hf = lane >> 4;
    const unsigned short* r = s + m * ld + 8 * hf;
    Frag f;
    f.h[0] = *(const v8us*)r;
    f.h[1] = *(const v8us*)(r + 16);
    return f.v;
}

// B fragment: 32x16 (K x N) with this lane's 32 K-values contiguous at srow.
__device__ __forceinline__ v16bf load_b_frag_kcontig(const unsigned short* srow) {
    int lane = threadIdx.x & 31;
    int hf = lane >> 4;
    const unsigned short* r = srow + 8 * hf;
    Frag f;
    f.h[0] = *(const v8us*)r;
    f.h[1] = *(const v8us*)(r + 16);
    return f.v;
}

// ---------------- f32 -> bf16 convert (row-major) ----------------
__global__ void cvt_bf16_kernel(const float* __restrict__ s, unsigned short* __restrict__ d, int n) {
    int i = blockIdx.x * 256 + threadIdx.x;
    if (i < n) d[i] = f2bf(s[i]);
}

// ---------------- f32 [K][N] -> bf16 transposed [N][K] ----------------
__global__ void cvtT_kernel(const float* __restrict__ src, unsigned short* __restrict__ dst,
                            int K, int N) {
    int idx = blockIdx.x * 256 + threadIdx.x;   // over N*K, k fastest -> coalesced writes
    if (idx >= N * K) return;
    int n = idx / K, k = idx - n * K;
    dst[idx] = f2bf(src[(size_t)k * N + n]);
}

// -------- WMMA GEMM: C[MxN] = A[MxK](bf16 row-major) @ BT[NxK](bf16), f32 out --------
// 128 threads = 4 waves; 64x64 block tile; double-buffered LDS filled by async DMA.
__global__ void __launch_bounds__(128) gemm_bf16_kernel(
        const unsigned short* __restrict__ A, const unsigned short* __restrict__ BT,
        float* __restrict__ Cm, int M, int K, int N) {
    __shared__ __align__(16) unsigned short sA[2][64 * 32];
    __shared__ __align__(16) unsigned short sB[2][64 * 32];
    int n0 = blockIdx.x * 64;
    int m0 = blockIdx.y * 64;
    int tid = threadIdx.x;
    int w = tid >> 5, lane = tid & 31;
    int hf = lane >> 4, nl = lane & 15;
    const unsigned short* Ab = A  + (size_t)m0 * K;
    const unsigned short* Bb = BT + (size_t)n0 * K;

    async_stage_tile<2, 4>(sA[0], Ab, K, tid, 128);      // 4 async b128 / thread / tile
    async_stage_tile<2, 4>(sB[0], Bb, K, tid, 128);

    v8f acc[4] = {};
    int buf = 0;
    for (int kk = 0; kk < K; kk += 32) {
        bool more = (kk + 32 < K);
        if (more) {                                      // DMA next tile behind compute
            async_stage_tile<2, 4>(sA[buf ^ 1], Ab + kk + 32, K, tid, 128);
            async_stage_tile<2, 4>(sB[buf ^ 1], Bb + kk + 32, K, tid, 128);
            wait_async<4>();                             // oldest 4 (current tile) done
        } else {
            wait_async<0>();
        }
        __syncthreads();                                 // tile visible to all waves
        v16bf a  = load_a_frag(sA[buf] + w * 16 * 32, 32);
        v16bf b0 = load_b_frag_kcontig(sB[buf] + (0 * 16 + nl) * 32);
        v16bf b1 = load_b_frag_kcontig(sB[buf] + (1 * 16 + nl) * 32);
        v16bf b2 = load_b_frag_kcontig(sB[buf] + (2 * 16 + nl) * 32);
        v16bf b3 = load_b_frag_kcontig(sB[buf] + (3 * 16 + nl) * 32);
        acc[0] = wmma_bf16(a, b0, acc[0]);
        acc[1] = wmma_bf16(a, b1, acc[1]);
        acc[2] = wmma_bf16(a, b2, acc[2]);
        acc[3] = wmma_bf16(a, b3, acc[3]);
        __syncthreads();                                 // all waves done with buf
        buf ^= 1;
    }
    int mb = m0 + w * 16;
#pragma unroll
    for (int nt = 0; nt < 4; ++nt)
#pragma unroll
        for (int j = 0; j < 8; ++j)
            Cm[(size_t)(mb + j + 8 * hf) * N + n0 + nt * 16 + nl] = acc[nt][j];
}

// ---------------- gate logits: glog[t][h] = x[t,:] . Wg[:,h] + bg[h] ----------------
__global__ void gate_kernel(const float* __restrict__ x, const float* __restrict__ Wg,
                            const float* __restrict__ bg, float* __restrict__ glog) {
    int idx = blockIdx.x * 256 + threadIdx.x;
    if (idx >= T_ * H_) return;
    int t = idx >> 4, h = idx & 15;
    float s = bg[h];
    for (int c = 0; c < C_; ++c) s = fmaf(x[(size_t)t * C_ + c], Wg[(size_t)c * H_ + h], s);
    glog[idx] = s;
}

// ---------------- log-sigmoid cumsum per head + input scale ----------------
__global__ void cumsum_kernel(const float* __restrict__ glog, float* __restrict__ la,
                              float* __restrict__ sig_in) {
    int h = threadIdx.x;
    if (h >= H_) return;
    float run = 0.f;
    for (int t = 0; t < T_; ++t) {
        float g = glog[t * H_ + h];
        float ls = fminf(g, 0.f) - log1pf(__expf(-fabsf(g)));  // log_sigmoid(g)
        run += ls;
        la[h * T_ + t] = run;
        sig_in[t * H_ + h] = 1.f / (1.f + __expf(g));          // sigmoid(-g)
    }
}

// ---------------- RoPE on q -> bf16 [H][T][DH] ----------------
__global__ void rope_q_kernel(const float* __restrict__ qf, unsigned short* __restrict__ qb) {
    int idx = blockIdx.x * 256 + threadIdx.x;   // (t, h, d<32)
    if (idx >= T_ * H_ * 32) return;
    int d = idx & 31;
    int h = (idx >> 5) & 15;
    int t = idx >> 9;
    float inv = powf(10000.f, -(float)(2 * d) / (float)DH_);
    float ang = (float)t * inv, sn, cs;
    __sincosf(ang, &sn, &cs);
    const float* row = qf + (size_t)t * (H_ * DH_) + h * DH_;
    float x1 = row[d], x2 = row[d + 32];
    unsigned short* o = qb + ((size_t)h * T_ + t) * DH_;
    o[d]      = f2bf(x1 * cs - x2 * sn);
    o[d + 32] = f2bf(x2 * cs + x1 * sn);
}

// --- RoPE on k; write k(bf16 [HKV][T][DH]), k_scaled(bf16 [H][T][DH]),
// --- v transposed (bf16 [HKV][DH][T]) so attention V tiles are K-contiguous.
__global__ void rope_kv_kernel(const float* __restrict__ kf, const float* __restrict__ vf,
                               const float* __restrict__ sig_in,
                               unsigned short* __restrict__ kb, unsigned short* __restrict__ ksb,
                               unsigned short* __restrict__ vbT) {
    int idx = blockIdx.x * 256 + threadIdx.x;   // (t, hkv, d<32)
    if (idx >= T_ * HKV_ * 32) return;
    int d = idx & 31;
    int hk = (idx >> 5) & 3;
    int t = idx >> 7;
    float inv = powf(10000.f, -(float)(2 * d) / (float)DH_);
    float ang = (float)t * inv, sn, cs;
    __sincosf(ang, &sn, &cs);
    const float* krow = kf + (size_t)t * (HKV_ * DH_) + hk * DH_;
    float k1 = krow[d], k2 = krow[d + 32];
    float r1 = k1 * cs - k2 * sn;
    float r2 = k2 * cs + k1 * sn;
    unsigned short* ko = kb + ((size_t)hk * T_ + t) * DH_;
    ko[d] = f2bf(r1); ko[d + 32] = f2bf(r2);
#pragma unroll
    for (int r = 0; r < 4; ++r) {
        int h = hk * 4 + r;
        float sc = sig_in[t * H_ + h];
        unsigned short* so = ksb + ((size_t)h * T_ + t) * DH_;
        so[d] = f2bf(r1 * sc); so[d + 32] = f2bf(r2 * sc);
    }
    const float* vrow = vf + (size_t)t * (HKV_ * DH_) + hk * DH_;
    vbT[((size_t)hk * DH_ + d)      * T_ + t] = f2bf(vrow[d]);
    vbT[((size_t)hk * DH_ + d + 32) * T_ + t] = f2bf(vrow[d + 32]);
}

// ---------------- gated-decay attention (flash-style) ----------------
// One wave per block; TDM fills double-buffered LDS tiles behind the WMMAs.
__global__ void __launch_bounds__(32) gdecay_attn_kernel(
        const unsigned short* __restrict__ qb, const unsigned short* __restrict__ ksb,
        const unsigned short* __restrict__ vbT, const float* __restrict__ la,
        float* __restrict__ yg) {
    __shared__ __align__(16) unsigned short sQ[16 * 64];
    __shared__ __align__(16) unsigned short sK[2][32 * 64];   // [key][dh] row-major
    __shared__ __align__(16) unsigned short sVT[2][64 * 32];  // [dh][key]
    __shared__ __align__(16) unsigned short sP[16 * 32];
    int h = blockIdx.y;
    int t0 = blockIdx.x * 16;
    int lane = threadIdx.x;
    int hf = lane >> 4, nl = lane & 15;
    const unsigned short* qh  = qb  + (size_t)h * T_ * DH_;
    const unsigned short* kh  = ksb + (size_t)h * T_ * DH_;
    const unsigned short* vhT = vbT + (size_t)(h >> 2) * DH_ * T_;
    const float* lah = la + (size_t)h * T_;

    tdm_load_2d(sQ,     qh + (size_t)t0 * DH_, 64, 16, 64);   // one TDM op per tile
    tdm_load_2d(sK[0],  kh,                    64, 32, 64);
    tdm_load_2d(sVT[0], vhT,                   32, 64, T_);
    float la_t[8];
#pragma unroll
    for (int j = 0; j < 8; ++j) la_t[j] = lah[t0 + j + 8 * hf];
    wait_tensor<2>();                                         // Q done (K,V in flight)
    v16bf aq0 = load_a_frag(sQ, 64);
    v16bf aq1 = load_a_frag(sQ + 32, 64);

    int smax = ((t0 + 15) >> 5) << 5;
    v8f y[4] = {};
    int buf = 0;
    for (int s0 = 0; s0 <= smax; s0 += 32) {
        if (s0 < smax) {                                      // TDM next block behind compute
            tdm_load_2d(sK[buf ^ 1],  kh + (size_t)(s0 + 32) * DH_, 64, 32, 64);
            tdm_load_2d(sVT[buf ^ 1], vhT + s0 + 32,                32, 64, T_);
            wait_tensor<2>();                                 // current block's 2 done (in order)
        } else {
            wait_tensor<0>();
        }
        float la_s[2] = { lah[s0 + nl], lah[s0 + 16 + nl] };
#pragma unroll
        for (int sub = 0; sub < 2; ++sub) {
            const unsigned short* krow = sK[buf] + (sub * 16 + nl) * 64;
            v16bf b0 = load_b_frag_kcontig(krow);
            v16bf b1 = load_b_frag_kcontig(krow + 32);
            v8f sc = {};
            sc = wmma_bf16(aq0, b0, sc);
            sc = wmma_bf16(aq1, b1, sc);
#pragma unroll
            for (int j = 0; j < 8; ++j) {
                int m = j + 8 * hf;
                int tq = t0 + m;
                int ts = s0 + sub * 16 + nl;
                float w = (ts <= tq) ? __expf(la_t[j] - la_s[sub]) : 0.f;
                sP[m * 32 + sub * 16 + nl] = f2bf(sc[j] * w);
            }
        }
        wait_lds();                                           // cross-lane P relayout
        v16bf pa  = load_a_frag(sP, 32);
        v16bf bv0 = load_b_frag_kcontig(sVT[buf] + (0 * 16 + nl) * 32);
        v16bf bv1 = load_b_frag_kcontig(sVT[buf] + (1 * 16 + nl) * 32);
        v16bf bv2 = load_b_frag_kcontig(sVT[buf] + (2 * 16 + nl) * 32);
        v16bf bv3 = load_b_frag_kcontig(sVT[buf] + (3 * 16 + nl) * 32);
        y[0] = wmma_bf16(pa, bv0, y[0]);
        y[1] = wmma_bf16(pa, bv1, y[1]);
        y[2] = wmma_bf16(pa, bv2, y[2]);
        y[3] = wmma_bf16(pa, bv3, y[3]);
        buf ^= 1;
    }
#pragma unroll
    for (int nt = 0; nt < 4; ++nt)
#pragma unroll
        for (int j = 0; j < 8; ++j)
            yg[(size_t)(t0 + j + 8 * hf) * C_ + h * DH_ + nt * 16 + nl] = y[nt][j];
}

// ---------------- local windowed softmax attention ----------------
__global__ void __launch_bounds__(256) local_attn_kernel(
        const unsigned short* __restrict__ qb, const unsigned short* __restrict__ kb,
        const unsigned short* __restrict__ vbT, float* __restrict__ yl) {
    __shared__ __align__(16) unsigned short sK[WIN_ * DH_];   // [key][dh]   16KB
    __shared__ __align__(16) unsigned short sVT[DH_ * WIN_];  // [dh][key]   16KB
    __shared__ __align__(16) unsigned short sQ[8][16 * DH_];  //             16KB
    __shared__ __align__(16) unsigned short sP[8][16 * 32];   //              8KB
    int h = blockIdx.y;
    int nb = blockIdx.x;
    int tid = threadIdx.x;
    int w = tid >> 5;            // wave id: query rows [w*16, w*16+15]
    int lane = tid & 31;
    int hf = lane >> 4, nl = lane & 15;
    int tbase = nb * WIN_;
    const unsigned short* qh  = qb  + (size_t)h * T_ * DH_;
    const unsigned short* kh  = kb  + (size_t)(h >> 2) * T_ * DH_;
    const unsigned short* vhT = vbT + (size_t)(h >> 2) * DH_ * T_;

    if (w == 0) {                                             // shared tiles: wave 0's TDM
        tdm_load_2d(sK,  kh + (size_t)tbase * DH_, 64, WIN_, 64);
        tdm_load_2d(sVT, vhT + tbase,              WIN_, 64, T_);
    }
    unsigned short* sQw = &sQ[w][0];
    tdm_load_2d(sQw, qh + (size_t)(tbase + w * 16) * DH_, 64, 16, 64);
    wait_tensor<0>();
    __syncthreads();

    v16bf aq0 = load_a_frag(sQw, 64);
    v16bf aq1 = load_a_frag(sQw + 32, 64);

    v8f sfr[8] = {};
    for (int kb2 = 0; kb2 <= w; ++kb2) {
        const unsigned short* krow = sK + (kb2 * 16 + nl) * 64;
        v16bf b0 = load_b_frag_kcontig(krow);
        v16bf b1 = load_b_frag_kcontig(krow + 32);
        v8f s = {};
        s = wmma_bf16(aq0, b0, s);
        s = wmma_bf16(aq1, b1, s);
        sfr[kb2] = s;
    }
    // softmax over keys (scale 1/sqrt(64)); row reductions across the 16-lane N group
    const float scale = 0.125f;
    float rmax[8], rsum[8];
#pragma unroll
    for (int j = 0; j < 8; ++j) rmax[j] = -1e30f;
    for (int kb2 = 0; kb2 <= w; ++kb2) {
#pragma unroll
        for (int j = 0; j < 8; ++j) {
            int m = j + 8 * hf, lq = w * 16 + m, lk = kb2 * 16 + nl;
            float v = (lk <= lq) ? sfr[kb2][j] * scale : -1e30f;
            rmax[j] = fmaxf(rmax[j], v);
        }
    }
#pragma unroll
    for (int j = 0; j < 8; ++j) {
        float v = rmax[j];
        for (int msk = 1; msk < 16; msk <<= 1) v = fmaxf(v, __shfl_xor(v, msk, 32));
        rmax[j] = v;
    }
#pragma unroll
    for (int j = 0; j < 8; ++j) rsum[j] = 0.f;
    for (int kb2 = 0; kb2 <= w; ++kb2) {
#pragma unroll
        for (int j = 0; j < 8; ++j) {
            int m = j + 8 * hf, lq = w * 16 + m, lk = kb2 * 16 + nl;
            float p = (lk <= lq) ? __expf(sfr[kb2][j] * scale - rmax[j]) : 0.f;
            sfr[kb2][j] = p;
            rsum[j] += p;
        }
    }
#pragma unroll
    for (int j = 0; j < 8; ++j) {
        float v = rsum[j];
        for (int msk = 1; msk < 16; msk <<= 1) v += __shfl_xor(v, msk, 32);
        rsum[j] = 1.f / v;
    }
    // P @ V in key-pairs of 32 via LDS re-layout
    unsigned short* sPw = &sP[w][0];
    v8f y[4] = {};
    for (int pr = 0; 2 * pr <= w; ++pr) {
#pragma unroll
        for (int sub = 0; sub < 2; ++sub) {
            int kb2 = 2 * pr + sub;
#pragma unroll
            for (int j = 0; j < 8; ++j) {
                int m = j + 8 * hf;
                float p = (kb2 <= w) ? sfr[kb2][j] * rsum[j] : 0.f;
                sPw[m * 32 + sub * 16 + nl] = f2bf(p);
            }
        }
        wait_lds();
        v16bf pa  = load_a_frag(sPw, 32);
        v16bf bv0 = load_b_frag_kcontig(sVT + (0 * 16 + nl) * WIN_ + pr * 32);
        v16bf bv1 = load_b_frag_kcontig(sVT + (1 * 16 + nl) * WIN_ + pr * 32);
        v16bf bv2 = load_b_frag_kcontig(sVT + (2 * 16 + nl) * WIN_ + pr * 32);
        v16bf bv3 = load_b_frag_kcontig(sVT + (3 * 16 + nl) * WIN_ + pr * 32);
        y[0] = wmma_bf16(pa, bv0, y[0]);
        y[1] = wmma_bf16(pa, bv1, y[1]);
        y[2] = wmma_bf16(pa, bv2, y[2]);
        y[3] = wmma_bf16(pa, bv3, y[3]);
    }
#pragma unroll
    for (int nt = 0; nt < 4; ++nt)
#pragma unroll
        for (int j = 0; j < 8; ++j)
            yl[(size_t)(tbase + w * 16 + j + 8 * hf) * C_ + h * DH_ + nt * 16 + nl] = y[nt][j];
}

// ---------------- fuse + RMSNorm -> bf16 ----------------
__global__ void __launch_bounds__(256) rmsnorm_kernel(
        const float* __restrict__ yg, const float* __restrict__ ylc,
        const float* __restrict__ wgt, unsigned short* __restrict__ out) {
    __shared__ float sY[C_];
    __shared__ float red[256];
    int t = blockIdx.x, tid = threadIdx.x;
    float ss = 0.f;
    for (int c = tid; c < C_; c += 256) {
        float v = yg[(size_t)t * C_ + c] + ylc[(size_t)t * C_ + c];
        sY[c] = v;
        ss += v * v;
    }
    red[tid] = ss;
    __syncthreads();
    for (int st = 128; st > 0; st >>= 1) {
        if (tid < st) red[tid] += red[tid + st];
        __syncthreads();
    }
    float r = rsqrtf(red[0] / (float)C_ + 1e-5f);
    for (int c = tid; c < C_; c += 256)
        out[(size_t)t * C_ + c] = f2bf(sY[c] * r * wgt[c]);
}

extern "C" void kernel_launch(void* const* d_in, const int* in_sizes, int n_in,
                              void* d_out, int out_size, void* d_ws, size_t ws_size,
                              hipStream_t stream) {
    (void)in_sizes; (void)n_in; (void)out_size; (void)ws_size;
    const float* x    = (const float*)d_in[0];
    const float* Wq   = (const float*)d_in[1];
    const float* Wk   = (const float*)d_in[2];
    const float* Wv   = (const float*)d_in[3];
    const float* Wc   = (const float*)d_in[4];
    const float* Wg   = (const float*)d_in[5];
    const float* bg   = (const float*)d_in[6];
    const float* rmsw = (const float*)d_in[7];
    float* out = (float*)d_out;

    char* p = (char*)d_ws;
    auto alloc = [&](size_t bytes) -> void* {
        void* r = (void*)p;
        p += (bytes + 255) & ~(size_t)255;
        return r;
    };
    unsigned short* xb   = (unsigned short*)alloc((size_t)T_ * C_ * 2);
    unsigned short* WqT  = (unsigned short*)alloc((size_t)C_ * C_ * 2);           // [N][K]
    unsigned short* WkT  = (unsigned short*)alloc((size_t)C_ * HKV_ * DH_ * 2);   // [N][K]
    unsigned short* WvT  = (unsigned short*)alloc((size_t)C_ * HKV_ * DH_ * 2);   // [N][K]
    unsigned short* WcT  = (unsigned short*)alloc((size_t)C_ * C_ * 2);           // [N][K]
    float* qf   = (float*)alloc((size_t)T_ * C_ * 4);
    float* kf   = (float*)alloc((size_t)T_ * HKV_ * DH_ * 4);
    float* vf   = (float*)alloc((size_t)T_ * HKV_ * DH_ * 4);
    float* glog = (float*)alloc((size_t)T_ * H_ * 4);
    float* la   = (float*)alloc((size_t)H_ * T_ * 4);
    float* sig  = (float*)alloc((size_t)T_ * H_ * 4);
    unsigned short* qbf  = (unsigned short*)alloc((size_t)H_ * T_ * DH_ * 2);
    unsigned short* kbf  = (unsigned short*)alloc((size_t)HKV_ * T_ * DH_ * 2);
    unsigned short* ksbf = (unsigned short*)alloc((size_t)H_ * T_ * DH_ * 2);
    unsigned short* vbfT = (unsigned short*)alloc((size_t)HKV_ * DH_ * T_ * 2);   // [HKV][DH][T]
    float* yg = (float*)alloc((size_t)T_ * C_ * 4);
    float* yl = (float*)alloc((size_t)T_ * C_ * 4);
    unsigned short* ybf = (unsigned short*)alloc((size_t)T_ * C_ * 2);

    int n;
    n = T_ * C_;         cvt_bf16_kernel<<<(n + 255) / 256, 256, 0, stream>>>(x, xb, n);
    n = C_ * C_;         cvtT_kernel<<<(n + 255) / 256, 256, 0, stream>>>(Wq, WqT, C_, C_);
    n = C_ * HKV_ * DH_; cvtT_kernel<<<(n + 255) / 256, 256, 0, stream>>>(Wk, WkT, C_, HKV_ * DH_);
    n = C_ * HKV_ * DH_; cvtT_kernel<<<(n + 255) / 256, 256, 0, stream>>>(Wv, WvT, C_, HKV_ * DH_);
    n = C_ * C_;         cvtT_kernel<<<(n + 255) / 256, 256, 0, stream>>>(Wc, WcT, C_, C_);

    gemm_bf16_kernel<<<dim3(C_ / 64, T_ / 64), 128, 0, stream>>>(xb, WqT, qf, T_, C_, C_);
    gemm_bf16_kernel<<<dim3((HKV_ * DH_) / 64, T_ / 64), 128, 0, stream>>>(xb, WkT, kf, T_, C_, HKV_ * DH_);
    gemm_bf16_kernel<<<dim3((HKV_ * DH_) / 64, T_ / 64), 128, 0, stream>>>(xb, WvT, vf, T_, C_, HKV_ * DH_);

    gate_kernel<<<(T_ * H_ + 255) / 256, 256, 0, stream>>>(x, Wg, bg, glog);
    cumsum_kernel<<<1, 32, 0, stream>>>(glog, la, sig);

    rope_q_kernel<<<(T_ * H_ * 32 + 255) / 256, 256, 0, stream>>>(qf, qbf);
    rope_kv_kernel<<<(T_ * HKV_ * 32 + 255) / 256, 256, 0, stream>>>(kf, vf, sig, kbf, ksbf, vbfT);

    gdecay_attn_kernel<<<dim3(T_ / 16, H_), 32, 0, stream>>>(qbf, ksbf, vbfT, la, yg);
    local_attn_kernel<<<dim3(T_ / WIN_, H_), 256, 0, stream>>>(qbf, kbf, vbfT, yl);

    rmsnorm_kernel<<<T_, 256, 0, stream>>>(yg, yl, rmsw, ybf);
    gemm_bf16_kernel<<<dim3(C_ / 64, T_ / 64), 128, 0, stream>>>(ybf, WcT, out, T_, C_, C_);
}
